// MultiHeadAttention_4458176053559
// MI455X (gfx1250) — matrix operand order repaired and checked by reading, not compiled, not run on hardware
//
#include <hip/hip_runtime.h>
#include <hip/hip_bf16.h>

// ---------------------------------------------------------------------------
// Multi-head attention, bf16 WMMA pipeline for gfx1250 (MI455X, wave32).
//   B=4, S=2048, E=1024, H=16, D=64
// Software-pipelined LDS staging: next tile's global loads are in flight
// while WMMA consumes the current tile; all transposes hoisted out of the
// hot loops (weights pre-transposed to [N,K], V written as [B,H,D,S]).
// ---------------------------------------------------------------------------

typedef __attribute__((ext_vector_type(16))) __bf16 v16bf;
typedef __attribute__((ext_vector_type(8)))  float  v8f;

#define EMBED    1024
#define SEQ      2048
#define BATCH    4
#define HEADS    16
#define HDIM     64
#define ROWS     (BATCH * SEQ)        // 8192

typedef unsigned short u16;
typedef unsigned int   u32;

// round-to-nearest-even fp32 -> bf16
static __device__ __forceinline__ u16 f2bf(float f) {
    u32 u = __float_as_uint(f);
    u32 r = (u + 0x7FFFu + ((u >> 16) & 1u)) >> 16;
    return (u16)r;
}

static __device__ __forceinline__ v8f wmma_bf16(v16bf a, v16bf b, v8f c) {
    return __builtin_amdgcn_wmma_f32_16x16x32_bf16(
        /*neg_a=*/false, a, /*neg_b=*/false, b,
        /*c_mod=*/(short)0, c, /*reuse_a=*/false, /*reuse_b=*/false);
}

// A-fragment (16x32 bf16, M x K). p points at [row0][k0] of a row-major
// [*, stride] bf16 array. ISA 7.12.2: lane = m + 16*half;
//   VGPR j<4 : k = 8*half + 2j, 2j+1 ; VGPR j>=4 : k = 16 + 8*half + 2(j-4)
static __device__ __forceinline__ v16bf load_frag_a(const u16* p, int stride) {
    const int lane = threadIdx.x & 31;
    const int m = lane & 15, lh = lane >> 4;
    const u16* rp = p + m * stride;
    union { v16bf v; u32 u[8]; } f;
#pragma unroll
    for (int j = 0; j < 8; ++j) {
        const int k = (j < 4) ? (8 * lh + 2 * j) : (16 + 8 * lh + 2 * (j - 4));
        f.u[j] = *(const u32*)(rp + k);
    }
    return f.v;
}

// B-fragment (32x16 bf16, K x N), stored TRANSPOSED in memory as [n][k].
// lane = n + 16*half ; VGPR j<4 : k = 8*half + 2j, 2j+1 ; VGPR j>=4 : k = 16 + 8*half + 2(j-4)
static __device__ __forceinline__ v16bf load_frag_b(const u16* p, int stride) {
    const int lane = threadIdx.x & 31;
    const int n = lane & 15, lh = lane >> 4;
    const u16* rp = p + n * stride;
    union { v16bf v; u32 u[8]; } f;
#pragma unroll
    for (int j = 0; j < 8; ++j) {
        const int k = (j < 4) ? (8 * lh + 2 * j) : (16 + 8 * lh + 2 * (j - 4));
        f.u[j] = *(const u32*)(rp + k);
    }
    return f.v;
}

// ---------------------------------------------------------------------------
// fp32 -> bf16 convert (plain) and convert+transpose [K,N] -> [N,K]
// ---------------------------------------------------------------------------
__global__ void cvt_f32_bf16(const float* __restrict__ src,
                             u16* __restrict__ dst, int n) {
    int i = blockIdx.x * blockDim.x + threadIdx.x;
    if (i < n) dst[i] = f2bf(src[i]);
}

__global__ void cvt_t_f32_bf16(const float* __restrict__ src,  // [K,N]
                               u16* __restrict__ dst,          // [N,K]
                               int K, int N) {
    int i = blockIdx.x * blockDim.x + threadIdx.x;
    if (i < N * K) {
        const int n = i / K, k = i - n * K;
        dst[i] = f2bf(src[(size_t)k * N + n]);
    }
}

// ---------------------------------------------------------------------------
// bf16 GEMM  C[M,N] = A[M,K] @ Bt^T + bias   (Bt is [N,K] row-major bf16)
// Block tile 128x128x32, 256 threads = 8 waves (2M x 4N), wave tile 64x32.
// Register-pipelined: next k-tile loads issue before WMMA on current tile.
// mode 0: qkv epilogue -> Q/K scatter to [B,H,S,D] (Q scaled 1/8),
//                         V scatter to [B,H,D,S] (pre-transposed for PV).
// mode 1: fp32 epilogue -> f_out[row*N + col]
// ---------------------------------------------------------------------------
__global__ __launch_bounds__(256)
void gemm_bf16_wmma(const u16* __restrict__ A, const u16* __restrict__ Bt,
                    const float* __restrict__ bias,
                    int M, int N, int K, int mode,
                    u16* __restrict__ q_out, u16* __restrict__ k_out,
                    u16* __restrict__ v_out, float* __restrict__ f_out) {
    constexpr int BM = 128, BN = 128, BK = 32, PAD = 8, LS = BK + PAD;
    __shared__ u16 sA[BM][LS];   // [m][k]
    __shared__ u16 sB[BN][LS];   // [n][k]

    const int tid  = threadIdx.x;
    const int lane = tid & 31, wave = tid >> 5;
    const int wm = wave >> 2, wn = wave & 3;
    const int m0 = blockIdx.y * BM, n0 = blockIdx.x * BN;

    // staging: thread -> (row = tid>>1, 16-elem column chunk = (tid&1)*16)
    const int r  = tid >> 1;
    const int cc = (tid & 1) * 16;
    const u16* gA = A  + (size_t)(m0 + r) * K + cc;
    const u16* gB = Bt + (size_t)(n0 + r) * K + cc;

    // preload tile 0 into registers
    uint4 ra0 = *(const uint4*)(gA);
    uint4 ra1 = *(const uint4*)(gA + 8);
    uint4 rb0 = *(const uint4*)(gB);
    uint4 rb1 = *(const uint4*)(gB + 8);

    v8f acc[4][2];
#pragma unroll
    for (int i = 0; i < 4; ++i)
#pragma unroll
        for (int j = 0; j < 2; ++j) acc[i][j] = (v8f){};

    for (int k0 = 0; k0 < K; k0 += BK) {
        // commit staged registers to LDS
        *(uint4*)&sA[r][cc]     = ra0;
        *(uint4*)&sA[r][cc + 8] = ra1;
        *(uint4*)&sB[r][cc]     = rb0;
        *(uint4*)&sB[r][cc + 8] = rb1;
        __syncthreads();

        // issue next tile's global loads (in flight during WMMA below)
        const bool more = (k0 + BK) < K;
        if (more) {
            ra0 = *(const uint4*)(gA + k0 + BK);
            ra1 = *(const uint4*)(gA + k0 + BK + 8);
            rb0 = *(const uint4*)(gB + k0 + BK);
            rb1 = *(const uint4*)(gB + k0 + BK + 8);
            if (k0 + 2 * BK < K) {  // gfx1250 global_prefetch of tile k0+2*BK
                __builtin_prefetch(gA + k0 + 2 * BK, 0, 1);
                __builtin_prefetch(gB + k0 + 2 * BK, 0, 1);
            }
        }

        // ---- WMMA core on current LDS tile ----
        v16bf af[4], bf[2];
#pragma unroll
        for (int mt = 0; mt < 4; ++mt)
            af[mt] = load_frag_a(&sA[wm * 64 + mt * 16][0], LS);
#pragma unroll
        for (int nt = 0; nt < 2; ++nt)
            bf[nt] = load_frag_b(&sB[wn * 32 + nt * 16][0], LS);
#pragma unroll
        for (int mt = 0; mt < 4; ++mt)
#pragma unroll
            for (int nt = 0; nt < 2; ++nt)
                acc[mt][nt] = wmma_bf16(af[mt], bf[nt], acc[mt][nt]);

        __syncthreads();   // everyone done reading before next commit
    }

    // ---- epilogue ----
    const int lh = lane >> 4, ln = lane & 15;
#pragma unroll
    for (int mt = 0; mt < 4; ++mt)
#pragma unroll
        for (int nt = 0; nt < 2; ++nt)
#pragma unroll
            for (int rr = 0; rr < 8; ++rr) {
                const int row = m0 + wm * 64 + mt * 16 + rr + 8 * lh;
                const int col = n0 + wn * 32 + nt * 16 + ln;
                float val = acc[mt][nt][rr] + bias[col];
                if (mode == 0) {
                    const int which = col >> 10;      // 0=q 1=k 2=v
                    const int e = col & 1023;
                    const int h = e >> 6, d = e & 63;
                    const int b = row >> 11, s = row & 2047;
                    if (which == 0) val *= 0.125f;    // fold 1/sqrt(64) into Q
                    const u16 bv = f2bf(val);
                    if (which == 2)                   // V transposed [B,H,D,S]
                        v_out[((size_t)(b * HEADS + h) * HDIM + d) * SEQ + s] = bv;
                    else if (which == 0)
                        q_out[((size_t)(b * HEADS + h) * SEQ + s) * HDIM + d] = bv;
                    else
                        k_out[((size_t)(b * HEADS + h) * SEQ + s) * HDIM + d] = bv;
                } else {
                    f_out[(size_t)row * N + col] = val;
                }
            }
}

// ---------------------------------------------------------------------------
// Flash attention: one workgroup per (b,h, 128-query block), 8 waves.
// Wave w owns 16 query rows. 64-key tiles, register-pipelined staging.
// K is [B,H,S,D]; V is pre-transposed [B,H,D,S] so both stages are b128 copies.
// ---------------------------------------------------------------------------
__global__ __launch_bounds__(256)
void attn_wmma(const u16* __restrict__ Qb, const u16* __restrict__ Kb,
               const u16* __restrict__ Vt, u16* __restrict__ Ob) {
    constexpr int KT = 64;            // keys per tile
    constexpr int NT = SEQ / KT;      // 32 tiles
    constexpr int KS = HDIM + 8;      // LDS stride (16B-aligned rows)
    __shared__ u16 sK[KT][KS];        // [key][d]  -> score B-frags
    __shared__ u16 sV[HDIM][KT + 8];  // [d][key]  -> PV B-frags
    __shared__ u16 sP[8][16 * 32];    // per-wave P staging (A-frag feed)

    const int blk  = blockIdx.x;
    const int qblk = blk & 15;
    const int bh   = blk >> 4;                  // b*16 + h
    const int tid  = threadIdx.x;
    const int lane = tid & 31, wave = tid >> 5;
    const int lh = lane >> 4, ln = lane & 15;

    const u16* Qp  = Qb + (size_t)bh * SEQ * HDIM;
    const u16* Kp  = Kb + (size_t)bh * SEQ * HDIM;
    const u16* Vtp = Vt + (size_t)bh * HDIM * SEQ;

    const int qrow0 = qblk * 128 + wave * 16;

    // Q fragments straight from global (rows contiguous in d)
    v16bf qf[2];
#pragma unroll
    for (int kk = 0; kk < 2; ++kk)
        qf[kk] = load_frag_a(Qp + (size_t)qrow0 * HDIM + kk * 32, HDIM);

    v8f o[4];
#pragma unroll
    for (int t = 0; t < 4; ++t) o[t] = (v8f){};
    float mr[8], lr[8];
#pragma unroll
    for (int r = 0; r < 8; ++r) { mr[r] = -3.0e38f; lr[r] = 0.f; }

    // staging geometry: 512 b128 chunks per tile per array, 2 per thread
    // K chunk: key = c>>3, d-chunk = (c&7)*8 ; V chunk: d = c>>3, key-chunk
    uint4 rk[2], rv[2];
#pragma unroll
    for (int pass = 0; pass < 2; ++pass) {
        const int c = tid + pass * 256;
        rk[pass] = *(const uint4*)(Kp  + (size_t)(c >> 3) * HDIM + (c & 7) * 8);
        rv[pass] = *(const uint4*)(Vtp + (size_t)(c >> 3) * SEQ  + (c & 7) * 8);
    }

    for (int kt = 0; kt < NT; ++kt) {
        // commit staged registers to LDS
#pragma unroll
        for (int pass = 0; pass < 2; ++pass) {
            const int c = tid + pass * 256;
            *(uint4*)&sK[c >> 3][(c & 7) * 8] = rk[pass];
            *(uint4*)&sV[c >> 3][(c & 7) * 8] = rv[pass];
        }
        __syncthreads();

        // issue next tile's global loads (overlap with compute below)
        if (kt + 1 < NT) {
#pragma unroll
            for (int pass = 0; pass < 2; ++pass) {
                const int c = tid + pass * 256;
                rk[pass] = *(const uint4*)(Kp + (size_t)((kt + 1) * KT + (c >> 3)) * HDIM
                                              + (c & 7) * 8);
                rv[pass] = *(const uint4*)(Vtp + (size_t)(c >> 3) * SEQ
                                               + (kt + 1) * KT + (c & 7) * 8);
            }
        }

        // ---- two 32-key chunks per tile ----
#pragma unroll
        for (int c2 = 0; c2 < 2; ++c2) {
            v8f s[2];
#pragma unroll
            for (int sub = 0; sub < 2; ++sub) {
                const int kb = c2 * 32 + sub * 16;
                v8f sc = (v8f){};
                sc = wmma_bf16(qf[0], load_frag_b(&sK[kb][0], KS), sc);
                sc = wmma_bf16(qf[1], load_frag_b(&sK[kb][32], KS), sc);
                s[sub] = sc;
            }
            // ---- online softmax over these 32 keys ----
#pragma unroll
            for (int r = 0; r < 8; ++r) {
                float v = fmaxf(s[0][r], s[1][r]);
#pragma unroll
                for (int msk = 1; msk < 16; msk <<= 1)
                    v = fmaxf(v, __shfl_xor(v, msk, 32));
                const float mnew = fmaxf(mr[r], v);
                const float rescale = __expf(mr[r] - mnew);
                mr[r] = mnew;
                lr[r] *= rescale;
#pragma unroll
                for (int t = 0; t < 4; ++t) o[t][r] *= rescale;
                const float p0 = __expf(s[0][r] - mnew);
                const float p1 = __expf(s[1][r] - mnew);
                s[0][r] = p0; s[1][r] = p1;
                float rs = p0 + p1;
#pragma unroll
                for (int msk = 1; msk < 16; msk <<= 1)
                    rs += __shfl_xor(rs, msk, 32);
                lr[r] += rs;
            }
            // ---- P (C-layout) -> per-wave LDS -> A-fragment ----
#pragma unroll
            for (int sub = 0; sub < 2; ++sub)
#pragma unroll
                for (int r = 0; r < 8; ++r)
                    sP[wave][(r + 8 * lh) * 32 + sub * 16 + ln] = f2bf(s[sub][r]);
            v16bf pf = load_frag_a(&sP[wave][0], 32);
            // ---- O += P @ V(32 x 64) ----
#pragma unroll
            for (int t = 0; t < 4; ++t)
                o[t] = wmma_bf16(pf, load_frag_b(&sV[t * 16][c2 * 32], KT + 8), o[t]);
        }
        __syncthreads();   // done reading sK/sV before next commit
    }

    // ---- normalize + write bf16 attention output [B,S,E] ----
    const int h = bh & 15, b = bh >> 4;
#pragma unroll
    for (int t = 0; t < 4; ++t)
#pragma unroll
        for (int r = 0; r < 8; ++r) {
            const int srow = qrow0 + r + 8 * lh;
            const int e = h * HDIM + t * 16 + ln;
            Ob[((size_t)(b * SEQ + srow)) * EMBED + e] = f2bf(o[t][r] / lr[r]);
        }
}

// ---------------------------------------------------------------------------
extern "C" void kernel_launch(void* const* d_in, const int* in_sizes, int n_in,
                              void* d_out, int out_size, void* d_ws, size_t ws_size,
                              hipStream_t stream) {
    const float* x     = (const float*)d_in[0];  // [4,2048,1024]
    const float* W_qkv = (const float*)d_in[1];  // [1024,3072]
    const float* b_qkv = (const float*)d_in[2];  // [3072]
    const float* W_out = (const float*)d_in[3];  // [1024,1024]
    const float* b_out = (const float*)d_in[4];  // [1024]
    float* out = (float*)d_out;                  // [4,2048,1024]

    char* ws = (char*)d_ws;
    size_t off = 0;
    u16* xb    = (u16*)(ws + off); off += (size_t)ROWS * EMBED * 2;        // 16 MB
    u16* wqkvt = (u16*)(ws + off); off += (size_t)3 * EMBED * EMBED * 2;   // 6 MB  [3072,1024]
    u16* woutt = (u16*)(ws + off); off += (size_t)EMBED * EMBED * 2;       // 2 MB  [1024,1024]
    u16* Qbuf  = (u16*)(ws + off); off += (size_t)ROWS * EMBED * 2;        // 16 MB [B,H,S,D]
    u16* Kbuf  = (u16*)(ws + off); off += (size_t)ROWS * EMBED * 2;        // 16 MB [B,H,S,D]
    u16* Vbuf  = (u16*)(ws + off); off += (size_t)ROWS * EMBED * 2;        // 16 MB [B,H,D,S]
    u16* Obuf  = (u16*)(ws + off); off += (size_t)ROWS * EMBED * 2;        // 16 MB [B,S,E]

    // 1) downconvert x; downconvert + transpose weights to [N,K]
    {
        int n0 = ROWS * EMBED;
        cvt_f32_bf16<<<(n0 + 255) / 256, 256, 0, stream>>>(x, xb, n0);
        int n1 = 3 * EMBED * EMBED;
        cvt_t_f32_bf16<<<(n1 + 255) / 256, 256, 0, stream>>>(W_qkv, wqkvt, EMBED, 3 * EMBED);
        int n2 = EMBED * EMBED;
        cvt_t_f32_bf16<<<(n2 + 255) / 256, 256, 0, stream>>>(W_out, woutt, EMBED, EMBED);
    }

    // 2) QKV projection (Q/K -> [B,H,S,D], V -> [B,H,D,S], Q pre-scaled 1/8)
    gemm_bf16_wmma<<<dim3((3 * EMBED) / 128, ROWS / 128), 256, 0, stream>>>(
        xb, wqkvt, b_qkv, ROWS, 3 * EMBED, EMBED, /*mode=*/0,
        Qbuf, Kbuf, Vbuf, nullptr);

    // 3) flash attention
    attn_wmma<<<BATCH * HEADS * (SEQ / 128), 256, 0, stream>>>(Qbuf, Kbuf, Vbuf, Obuf);

    // 4) output projection -> fp32 d_out
    gemm_bf16_wmma<<<dim3(EMBED / 128, ROWS / 128), 256, 0, stream>>>(
        Obuf, woutt, b_out, ROWS, EMBED, EMBED, /*mode=*/1,
        nullptr, nullptr, nullptr, out);
}
